// MinGRUModel_82162724372990
// MI455X (gfx1250) — compile-verified, matrix operand order checked by reading, not run on hardware
//
#include <hip/hip_runtime.h>
#include <hip/hip_bf16.h>

// ---------------------------------------------------------------------------
// MinGRU fused kernel for MI455X (gfx1250, wave32, WMMA + TDM gather).
//
//  h_t = sigmoid(-g_t)*h_{t-1} + sigmoid(g_t)*G(x_t),  G(x)=x+0.5 | sigmoid(x)
//  (exactly the reference's log-space scan, evaluated in linear space)
//
//  Per wave: 16 hidden + 16 gate channels, scans all L=2048 timesteps.
//  A-tiles (16 timesteps x 512 bf16) are gathered by the Tensor Data Mover
//  (gather mode, 16 row indices = the tokens) directly into LDS — no VALU
//  staging — double-buffered and overlapped with the WMMA loop.
// ---------------------------------------------------------------------------

typedef __attribute__((ext_vector_type(16))) __bf16 v16bf;
typedef __attribute__((ext_vector_type(8)))  float  v8f;
typedef unsigned int v4u __attribute__((ext_vector_type(4)));
typedef int          v8i __attribute__((ext_vector_type(8)));
typedef int          v4i __attribute__((ext_vector_type(4)));

#define B_SZ   64
#define L_SZ   2048
#define V_SZ   4096
#define E_SZ   512
#define TWO_E  1024
#define NCG    32          // channel groups of 16 (E/16)
#define CHUNK  16          // timesteps per WMMA M-tile
#define KCN    16          // K-chunks of 32 (E/32)
#define NCHUNK (L_SZ / CHUNK)

// round-to-nearest-even f32 -> bf16 bits
__device__ __forceinline__ unsigned short f2bf(float f) {
    unsigned int u = __float_as_uint(f);
    unsigned int r = u + 0x7FFFu + ((u >> 16) & 1u);
    return (unsigned short)(r >> 16);
}

// 16-bit A/B fragment k-offset: k = (j&7) + 8*(lane>=16) + 16*(j>=8)
__device__ __forceinline__ int koff(int lane, int j) {
    return (j & 7) + (((lane >> 4) & 1) << 3) + ((j >> 3) << 4);
}

struct a32 { uint4 lo, hi; };
union frag_u { a32 q; v16bf v; };

// ---- one-time: emb (f32) -> emb_bf (bf16), pairwise packed
__global__ void emb_prep_kernel(const float* __restrict__ emb,
                                unsigned int* __restrict__ emb_bf_u32) {
    const int i = blockIdx.x * blockDim.x + threadIdx.x;   // f32 pair index
    if (i < V_SZ * E_SZ / 2) {
        const float x0 = emb[2 * i + 0];
        const float x1 = emb[2 * i + 1];
        emb_bf_u32[i] = (unsigned int)f2bf(x0) | ((unsigned int)f2bf(x1) << 16);
    }
}

__global__ __launch_bounds__(32)
void mingru_fused_kernel(const long long* __restrict__ tokens,
                         const float* __restrict__ w_hg,
                         const float* __restrict__ w_fc,
                         const unsigned short* __restrict__ emb_bf,
                         float* __restrict__ partial) {
    __shared__ alignas(32) unsigned short sW[KCN * 2 * 32 * 16];   // 32 KB B-frags
    __shared__ alignas(32) unsigned short sA[2][CHUNK * E_SZ];     // 2 x 16 KB A tiles

    const int tid = threadIdx.x;           // 0..31 (one wave)
    const int b   = blockIdx.x;            // batch
    const int cg  = blockIdx.y;            // channel group
    const int f0  = cg * 16;

    // ---- one-time: stage w_hg slice as bf16 B-fragments (pre-swizzled)
    for (int kc = 0; kc < KCN; ++kc) {
        for (int t2 = 0; t2 < 2; ++t2) {
            const int col = (t2 ? E_SZ : 0) + f0 + (tid & 15);
            #pragma unroll
            for (int j2 = 0; j2 < 8; ++j2) {
                const int j = 2 * j2;
                const int k = kc * 32 + koff(tid, j);
                const float w0 = w_hg[(k + 0) * TWO_E + col];
                const float w1 = w_hg[(k + 1) * TWO_E + col];
                *(unsigned int*)&sW[((kc * 2 + t2) * 32 + tid) * 16 + j] =
                    (unsigned int)f2bf(w0) | ((unsigned int)f2bf(w1) << 16);
            }
        }
    }
    __syncthreads();

    const unsigned long long ga = (unsigned long long)(uintptr_t)emb_bf;
    const unsigned lds_a[2] = { (unsigned)(uintptr_t)&sA[0][0],
                                (unsigned)(uintptr_t)&sA[1][0] };

    // ---- TDM gather: 16 token-indexed bf16 rows (1 KB each) -> LDS tile
    auto issue_tile = [&](int cidx, unsigned ldsa) {
        const int base = b * L_SZ + cidx * CHUNK;
        v4i g2, g3;                               // 16-bit row indices
        #pragma unroll
        for (int p = 0; p < 4; ++p) {
            const unsigned a0 = (unsigned)tokens[base + 2 * p + 0] & 0xFFFFu;
            const unsigned a1 = (unsigned)tokens[base + 2 * p + 1] & 0xFFFFu;
            const unsigned b0 = (unsigned)tokens[base + 8 + 2 * p + 0] & 0xFFFFu;
            const unsigned b1 = (unsigned)tokens[base + 8 + 2 * p + 1] & 0xFFFFu;
            g2[p] = (int)(a0 | (a1 << 16));
            g3[p] = (int)(b0 | (b1 << 16));
        }
        v4u g0;
        g0[0] = 0x80000001u;                          // gather_mode=1, 16b idx, count=1
        g0[1] = ldsa;                                 // lds_addr
        g0[2] = (unsigned)(ga & 0xFFFFFFFFu);         // global_addr[31:0]
        g0[3] = (unsigned)((ga >> 32) & 0x01FFFFFFu)  // global_addr[56:32]
                | 0x80000000u;                        // type=2 ("image")
        v8i g1;
        g1[0] = (int)(1u << 16);                      // data_size=2B, wg_mask=0
        g1[1] = (int)((unsigned)(E_SZ & 0xFFFF) << 16);        // tensor_dim0=512
        g1[2] = (int)((unsigned)(V_SZ & 0xFFFF) << 16);        // tensor_dim1=4096
        g1[3] = (int)((unsigned)(E_SZ & 0xFFFF) << 16);        // tile_dim0=512
        g1[4] = CHUNK;                                // tile_dim1 = 16 valid indices
        g1[5] = E_SZ;                                 // tensor_dim0_stride = 512
        g1[6] = 0;
        g1[7] = 0;
        asm volatile("" ::: "memory");                // keep LDS reads before re-fill
#if __clang_major__ >= 23
        v8i g4 = {};
        __builtin_amdgcn_tensor_load_to_lds(g0, g1, g2, g3, g4, 0);
#else
        __builtin_amdgcn_tensor_load_to_lds(g0, g1, g2, g3, 0);
#endif
    };

    issue_tile(0, lds_a[0]);

    float h = 0.0f;  // recurrence state for channel f0 + (tid & 15)

    for (int c = 0; c < NCHUNK; ++c) {
        __builtin_amdgcn_s_wait_tensorcnt(0);      // current tile resident
        __syncthreads();
        if (c + 1 < NCHUNK)
            issue_tile(c + 1, lds_a[(c + 1) & 1]); // overlap next DMA with WMMA

        // ---- GEMM tile from row-major bf16 LDS: lane reads (m = lane&15) row
        const unsigned short* arow = &sA[c & 1][(tid & 15) * E_SZ];
        const int kshift = (tid >> 4) << 3;        // +8 k for high lane half
        v8f accH = {};
        v8f accG = {};
        #pragma unroll
        for (int kc = 0; kc < KCN; ++kc) {
            const int kb = kc * 32 + kshift;
            frag_u ua;
            ua.q.lo = *(const uint4*)&arow[kb];        // j=0..7  -> k = kb..kb+7
            ua.q.hi = *(const uint4*)&arow[kb + 16];   // j=8..15 -> k = kb+16..kb+23
            const v16bf wh = *(const v16bf*)&sW[((kc * 2 + 0) * 32 + tid) * 16];
            const v16bf wg = *(const v16bf*)&sW[((kc * 2 + 1) * 32 + tid) * 16];
            accH = __builtin_amdgcn_wmma_f32_16x16x32_bf16(
                       false, ua.v, false, wh, (short)0, accH, false, false);
            accG = __builtin_amdgcn_wmma_f32_16x16x32_bf16(
                       false, ua.v, false, wg, (short)0, accG, false, false);
        }

        // ---- sequential scan over 16 timesteps (lane pair shares a channel)
        const bool hi = (tid >= 16);
        float Ht[16], Gt[16];
        #pragma unroll
        for (int r = 0; r < 8; ++r) {
            const float oh = __shfl_xor(accH[r], 16, 32);
            const float og = __shfl_xor(accG[r], 16, 32);
            Ht[r]     = hi ? oh      : accH[r];
            Gt[r]     = hi ? og      : accG[r];
            Ht[r + 8] = hi ? accH[r] : oh;
            Gt[r + 8] = hi ? accG[r] : og;
        }
        #pragma unroll
        for (int t = 0; t < 16; ++t) {
            const float g = Gt[t];
            const float x = Ht[t];
            const float z  = 1.0f / (1.0f + __expf(-g));          // sigmoid(g)
            const float gx = (x >= 0.0f) ? (x + 0.5f)
                                         : 1.0f / (1.0f + __expf(-x));
            h = fmaf(1.0f - z, h, z * gx);                        // (1-z)=sigmoid(-g)
        }
    }

    // ---- deterministic partial: h_last . w_fc (channels duplicated in pair)
    float contrib = h * w_fc[f0 + (tid & 15)];
    #pragma unroll
    for (int off = 16; off > 0; off >>= 1)
        contrib += __shfl_xor(contrib, off, 32);
    if (tid == 0)
        partial[b * NCG + cg] = 0.5f * contrib;
}

__global__ void mingru_reduce_kernel(const float* __restrict__ partial,
                                     const float* __restrict__ b_fc,
                                     float* __restrict__ out) {
    const int b = threadIdx.x;
    if (b < B_SZ) {
        float s = b_fc[0];
        #pragma unroll
        for (int i = 0; i < NCG; ++i) s += partial[b * NCG + i];
        out[b] = s;
    }
}

extern "C" void kernel_launch(void* const* d_in, const int* in_sizes, int n_in,
                              void* d_out, int out_size, void* d_ws, size_t ws_size,
                              hipStream_t stream) {
    const long long* tokens = (const long long*)d_in[0]; // (64, 2048) int64
    const float*     emb    = (const float*)d_in[1];     // (4096, 512)
    const float*     w_hg   = (const float*)d_in[2];     // (512, 1024)
    const float*     w_fc   = (const float*)d_in[3];     // (1, 512)
    const float*     b_fc   = (const float*)d_in[4];     // (1,)
    float* out = (float*)d_out;                          // (64, 1)

    float*          partial = (float*)d_ws;                                  // 8 KB
    unsigned short* emb_bf  = (unsigned short*)((char*)d_ws + 8192);         // 4 MB

    emb_prep_kernel<<<(V_SZ * E_SZ / 2 + 255) / 256, 256, 0, stream>>>(
        emb, (unsigned int*)emb_bf);

    dim3 grid(B_SZ, NCG);
    mingru_fused_kernel<<<grid, 32, 0, stream>>>(tokens, w_hg, w_fc, emb_bf, partial);
    mingru_reduce_kernel<<<1, 64, 0, stream>>>(partial, b_fc, out);
}
